// NodeAttention_45956150067207
// MI455X (gfx1250) — matrix operand order repaired and checked by reading, not compiled, not run on hardware
//
#include <hip/hip_runtime.h>
#include <hip/hip_bf16.h>

#define N_ 1024
#define D_ 128
#define H_ 8
#define DH_ 32
#define INNER_ 256
#define NN_ (N_*N_)

typedef _Float16 v16h __attribute__((ext_vector_type(16)));
typedef _Float16 v8h  __attribute__((ext_vector_type(8)));
typedef float    v8f  __attribute__((ext_vector_type(8)));
typedef float    v4f  __attribute__((ext_vector_type(4)));
typedef float    v2f  __attribute__((ext_vector_type(2)));

__device__ __forceinline__ v8f wmma16x16x32(v16h a, v16h b, v8f c) {
  // D(16x16 f32) = A(16x32 f16) x B(32x16 f16) + C
  return __builtin_amdgcn_wmma_f32_16x16x32_f16(false, a, false, b, (short)0, c, false, false);
}

// ---- cross-lane helpers on VALU (v_permlane16 / v_permlanex16), no DS waits ----
__device__ __forceinline__ float swap16_f(float v) {
  // exchange with lane^16 (identity select into the other 16-lane row)
  unsigned int x = __float_as_uint(v);
  unsigned int y = __builtin_amdgcn_permlanex16(x, x, 0x76543210u, 0xFEDCBA98u, false, false);
  return __uint_as_float(y);
}
__device__ __forceinline__ float perm16_f(float v, unsigned int lo, unsigned int hi) {
  unsigned int x = __float_as_uint(v);
  unsigned int y = __builtin_amdgcn_permlane16(x, x, lo, hi, false, false);
  return __uint_as_float(y);
}
// max-reduce within each 16-lane group (xor 1,2,4,8)
__device__ __forceinline__ float grp16_max(float v) {
  v = fmaxf(v, perm16_f(v, 0x67452301u, 0xEFCDAB89u));  // xor 1
  v = fmaxf(v, perm16_f(v, 0x45670123u, 0xCDEF89ABu));  // xor 2
  v = fmaxf(v, perm16_f(v, 0x01234567u, 0x89ABCDEFu));  // xor 4
  v = fmaxf(v, perm16_f(v, 0xFEDCBA98u, 0x76543210u));  // xor 8
  return v;
}

// A-matrix 16x32 f16 layout (ISA table): lane m(0..15)/m+16 holds row m.
// halves 0..7 -> K = hi*8 + u ; halves 8..15 -> K = 16 + hi*8 + u
__device__ __forceinline__ v16h ldA16x32(const _Float16* base, int row0, int rowStride,
                                         int col0, int m, int hi) {
  const _Float16* p = base + (size_t)(row0 + m) * rowStride + col0 + hi * 8;
  v8h a = *(const v8h*)p;
  v8h b = *(const v8h*)(p + 16);
  v16h t;
#pragma unroll
  for (int u = 0; u < 8; ++u) { t[u] = a[u]; t[u + 8] = b[u]; }
  return t;
}

// B-matrix 32x16 f16 layout: lane n(0..15)/n+16 holds column n;
// halves 0..7 -> K = hi*8 + u ; halves 8..15 -> K = 16 + hi*8 + u (same as A).
__device__ __forceinline__ v16h ldB32x16(const _Float16* base, int row0, int rowStride,
                                         int col0, int m, int hi) {
  const _Float16* p = base + (size_t)(row0 + m) * rowStride + col0 + hi * 8;
  v8h a = *(const v8h*)p;
  v8h b = *(const v8h*)(p + 16);
  v16h t;
#pragma unroll
  for (int u = 0; u < 8; ++u) { t[u] = a[u]; t[u + 8] = b[u]; }
  return t;
}

// ---------------- node LayerNorm ----------------
__global__ __launch_bounds__(128) void ln_node_kernel(const float* __restrict__ x,
                                                      const float* __restrict__ w,
                                                      const float* __restrict__ b,
                                                      float* __restrict__ xn) {
  int row = blockIdx.x, tid = threadIdx.x;
  float v = x[row * D_ + tid];
  float s = v, ss = v * v;
#pragma unroll
  for (int mk = 16; mk >= 1; mk >>= 1) { s += __shfl_xor(s, mk); ss += __shfl_xor(ss, mk); }
  __shared__ float s1[4], s2[4];
  if ((tid & 31) == 0) { s1[tid >> 5] = s; s2[tid >> 5] = ss; }
  __syncthreads();
  float S = s1[0] + s1[1] + s1[2] + s1[3];
  float SS = s2[0] + s2[1] + s2[2] + s2[3];
  float mu = S * (1.0f / D_);
  float var = SS * (1.0f / D_) - mu * mu;
  float rstd = __builtin_amdgcn_rsqf(var + 1e-5f);
  xn[row * D_ + tid] = (v - mu) * rstd * w[tid] + b[tid];
}

// ---------------- Wbias prep: W2t[n][d] = w_d*Wb[d,n] (f16, n<8 else 0), C0 ----------------
__global__ __launch_bounds__(128) void prep_wbias_kernel(const float* __restrict__ pnw,
                                                         const float* __restrict__ pnb,
                                                         const float* __restrict__ Wb,
                                                         _Float16* __restrict__ W2t,
                                                         float* __restrict__ C0) {
  int d = threadIdx.x;
  for (int n = 0; n < 16; ++n)
    W2t[n * D_ + d] = (n < H_) ? (_Float16)(pnw[d] * Wb[d * H_ + n]) : (_Float16)0.0f;
  if (d < H_) {
    float c0 = 0.f;
    for (int k = 0; k < D_; ++k) c0 += pnb[k] * Wb[k * H_ + d];
    C0[d] = c0;
  }
}

// ---------------- qkvg = xn @ Wqkv, store q/k f16 (n,dh), v f16 dh-major, sigmoid(g) f32 ----------------
__global__ __launch_bounds__(256) void qkvg_kernel(const float* __restrict__ xn,
                                                   const float* __restrict__ Wqkv,
                                                   _Float16* __restrict__ qh,
                                                   _Float16* __restrict__ kh,
                                                   _Float16* __restrict__ vT,
                                                   float* __restrict__ sg) {
  __shared__ float xs[4][D_];
  int r0 = blockIdx.x * 4;
  int c0 = blockIdx.y * 256;
  int tid = threadIdx.x;
  for (int t = tid; t < 4 * D_; t += 256) xs[t >> 7][t & 127] = xn[(r0 + (t >> 7)) * D_ + (t & 127)];
  __syncthreads();
  int c = c0 + tid;
  float a0 = 0.f, a1 = 0.f, a2 = 0.f, a3 = 0.f;
  for (int d = 0; d < D_; ++d) {
    float wv = Wqkv[d * 1024 + c];
    a0 = fmaf(xs[0][d], wv, a0);
    a1 = fmaf(xs[1][d], wv, a1);
    a2 = fmaf(xs[2][d], wv, a2);
    a3 = fmaf(xs[3][d], wv, a3);
  }
  int part = c >> 8, cw = c & 255, hh = cw >> 5, dh = cw & 31;
  float acc[4] = {a0, a1, a2, a3};
#pragma unroll
  for (int k = 0; k < 4; ++k) {
    int n = r0 + k;
    float a = acc[k];
    if (part == 0)      qh[((size_t)hh * N_ + n) * DH_ + dh] = (_Float16)a;
    else if (part == 1) kh[((size_t)hh * N_ + n) * DH_ + dh] = (_Float16)a;
    else if (part == 2) vT[((size_t)hh * DH_ + dh) * N_ + n] = (_Float16)a;
    else                sg[((size_t)hh * N_ + n) * DH_ + dh] = 1.0f / (1.0f + __expf(-a));
  }
}

// ---------------- fused pair-LN + bias projection + mask (the 512MB streamer) ----------------
__global__ __launch_bounds__(256) void bias_pair_kernel(const float* __restrict__ pf,
                                                        const int* __restrict__ mask,
                                                        const _Float16* __restrict__ W2t,
                                                        const float* __restrict__ C0,
                                                        float* __restrict__ biasM) {
  int lane = threadIdx.x & 31;
  int wid = (int)((blockIdx.x * blockDim.x + threadIdx.x) >> 5);
  int nwaves = (int)((gridDim.x * blockDim.x) >> 5);
  int m = lane & 15, hi = lane >> 4;

  // Preload B chunks (w ⊙ Wbias, cols 8..15 zero), once per wave.
  v16h Bm[4];
#pragma unroll
  for (int kc = 0; kc < 4; ++kc) Bm[kc] = ldB32x16(W2t, 0, D_, kc * 32, m, hi);
  float c0 = C0[m & 7];

  const int TILES = NN_ / 16;
  for (int tile = wid; tile < TILES; tile += nwaves) {
    int p0 = tile * 16;
    const float* row = pf + (size_t)(p0 + m) * D_;
    float xr[64];
#pragma unroll
    for (int kc = 0; kc < 4; ++kc) {
      int dA = kc * 32 + hi * 8;
      int dB = dA + 16;
      // read-once 512MB stream: non-temporal so it doesn't evict biasM/q/k/v from L2
      *(v4f*)&xr[kc * 16 + 0]  = __builtin_nontemporal_load((const v4f*)(row + dA));
      *(v4f*)&xr[kc * 16 + 4]  = __builtin_nontemporal_load((const v4f*)(row + dA + 4));
      *(v4f*)&xr[kc * 16 + 8]  = __builtin_nontemporal_load((const v4f*)(row + dB));
      *(v4f*)&xr[kc * 16 + 12] = __builtin_nontemporal_load((const v4f*)(row + dB + 4));
    }
    // packed (v_pk_*) 2-wide accumulators, 2x ILP each, to break the serial FP chain
    v2f sA = {0.f, 0.f}, sB = {0.f, 0.f}, qA = {0.f, 0.f}, qB = {0.f, 0.f};
#pragma unroll
    for (int u = 0; u < 64; u += 4) {
      v2f x0 = {xr[u + 0], xr[u + 1]};
      v2f x1 = {xr[u + 2], xr[u + 3]};
      sA += x0; qA += x0 * x0;
      sB += x1; qB += x1 * x1;
    }
    float s  = (sA.x + sA.y) + (sB.x + sB.y);
    float ss = (qA.x + qA.y) + (qB.x + qB.y);
    s  += swap16_f(s);    // combine the two half-row partials (lane ^ 16)
    ss += swap16_f(ss);
    float mu = s * (1.0f / D_);
    float var = ss * (1.0f / D_) - mu * mu;
    float rstd = __builtin_amdgcn_rsqf(var + 1e-5f);
    float nm = -mu * rstd;

    // normalize in-register (lane layout == A row layout), then f16 A tiles
    v8f acc = {0.f, 0.f, 0.f, 0.f, 0.f, 0.f, 0.f, 0.f};
#pragma unroll
    for (int kc = 0; kc < 4; ++kc) {
      v16h a;
#pragma unroll
      for (int u = 0; u < 16; ++u) a[u] = (_Float16)fmaf(xr[kc * 16 + u], rstd, nm);
      acc = wmma16x16x32(a, Bm[kc], acc);
    }
    // epilogue: bias_h = acc + C0_h ; fold mask -> -1e30 (no cross-lane traffic)
#pragma unroll
    for (int r = 0; r < 8; ++r) {
      int p = p0 + r + (hi << 3);
      float bh = acc[r] + c0;
      int mv = mask[p];
      float outv = mv ? bh : -1e30f;
      if (m < H_) biasM[(size_t)m * NN_ + p] = outv;
    }
  }
}

// ---------------- WMMA flash attention: one wave per (head, 16-row tile) ----------------
__global__ __launch_bounds__(256) void attn_kernel(const _Float16* __restrict__ qh,
                                                   const _Float16* __restrict__ kh,
                                                   const _Float16* __restrict__ vT,
                                                   const float* __restrict__ sg,
                                                   const float* __restrict__ biasM,
                                                   float* __restrict__ attn_out) {
  __shared__ __align__(16) _Float16 plds[8][16 * 32];
  int lane = threadIdx.x & 31;
  int w = threadIdx.x >> 5;
  int wid = blockIdx.x * 8 + w;          // 512 waves total
  int h = wid >> 6;                      // 64 i-tiles per head
  int i0 = (wid & 63) << 4;
  int m = lane & 15, hi = lane >> 4;

  const _Float16* qbase = qh + (size_t)h * N_ * DH_;
  const _Float16* kbase = kh + (size_t)h * N_ * DH_;
  const _Float16* vbase = vT + (size_t)h * DH_ * N_;

  v16h aq = ldA16x32(qbase, i0, DH_, 0, m, hi);
  v16h bones;
#pragma unroll
  for (int u = 0; u < 16; ++u) bones[u] = (_Float16)1.0f;

  v8f o0   = {0.f,0.f,0.f,0.f,0.f,0.f,0.f,0.f};
  v8f o1   = {0.f,0.f,0.f,0.f,0.f,0.f,0.f,0.f};
  v8f lacc = {0.f,0.f,0.f,0.f,0.f,0.f,0.f,0.f};  // softmax denominators (row sums)
  float mrun[8];
#pragma unroll
  for (int r = 0; r < 8; ++r) mrun[r] = -3.0e38f;
  const float scale = 0.17677669529663689f;  // 1/sqrt(32)

  for (int j0 = 0; j0 < N_; j0 += 32) {
    v16h bk0 = ldB32x16(kbase, j0,      DH_, 0, m, hi);
    v16h bk1 = ldB32x16(kbase, j0 + 16, DH_, 0, m, hi);
    v8f z = {0.f,0.f,0.f,0.f,0.f,0.f,0.f,0.f};
    v8f s0 = wmma16x16x32(aq, bk0, z);
    v8f s1 = wmma16x16x32(aq, bk1, z);

    const float* bb = biasM + (size_t)h * NN_ + (size_t)(i0 + (hi << 3)) * N_ + j0 + m;
    float l0[8], l1[8], mnew[8];
#pragma unroll
    for (int r = 0; r < 8; ++r) {
      l0[r] = s0[r] * scale + bb[(size_t)r * N_];
      l1[r] = s1[r] * scale + bb[(size_t)r * N_ + 16];
      float t = grp16_max(fmaxf(l0[r], l1[r]));     // VALU-only row max
      mnew[r] = fmaxf(mrun[r], t);
      float corr = __expf(mrun[r] - mnew[r]);
      mrun[r] = mnew[r];
      lacc[r] *= corr;
      o0[r] *= corr;
      o1[r] *= corr;
    }
#pragma unroll
    for (int r = 0; r < 8; ++r) {
      float p0f = __expf(l0[r] - mnew[r]);
      float p1f = __expf(l1[r] - mnew[r]);
      int rowm = r + (hi << 3);
      plds[w][rowm * 32 + m]      = (_Float16)p0f;
      plds[w][rowm * 32 + 16 + m] = (_Float16)p1f;
    }
    v16h ap  = ldA16x32(&plds[w][0], 0, 32, 0, m, hi);
    v16h bv0 = ldB32x16(vbase, 0,  N_, j0, m, hi);
    v16h bv1 = ldB32x16(vbase, 16, N_, j0, m, hi);
    lacc = wmma16x16x32(ap, bones, lacc);   // row sums land in C/D layout (all cols equal)
    o0 = wmma16x16x32(ap, bv0, o0);
    o1 = wmma16x16x32(ap, bv1, o1);
  }

#pragma unroll
  for (int r = 0; r < 8; ++r) {
    int row = i0 + r + (hi << 3);
    float inv = 1.0f / lacc[r];
    float g0 = sg[((size_t)h * N_ + row) * DH_ + m];
    float g1 = sg[((size_t)h * N_ + row) * DH_ + 16 + m];
    attn_out[(size_t)row * INNER_ + h * DH_ + m]      = o0[r] * inv * g0;
    attn_out[(size_t)row * INNER_ + h * DH_ + 16 + m] = o1[r] * inv * g1;
  }
}

// ---------------- output projection ----------------
__global__ __launch_bounds__(128) void outproj_kernel(const float* __restrict__ attn,
                                                      const float* __restrict__ Wout,
                                                      const float* __restrict__ bout,
                                                      float* __restrict__ out) {
  __shared__ float xs[4][INNER_];
  int r0 = blockIdx.x * 4;
  int tid = threadIdx.x;
  for (int t = tid; t < 4 * INNER_; t += 128)
    xs[t >> 8][t & 255] = attn[(size_t)(r0 + (t >> 8)) * INNER_ + (t & 255)];
  __syncthreads();
  float a0 = 0.f, a1 = 0.f, a2 = 0.f, a3 = 0.f;
  for (int d = 0; d < INNER_; ++d) {
    float wv = Wout[d * D_ + tid];
    a0 = fmaf(xs[0][d], wv, a0);
    a1 = fmaf(xs[1][d], wv, a1);
    a2 = fmaf(xs[2][d], wv, a2);
    a3 = fmaf(xs[3][d], wv, a3);
  }
  float bv = bout[tid];
  out[(size_t)(r0 + 0) * D_ + tid] = a0 + bv;
  out[(size_t)(r0 + 1) * D_ + tid] = a1 + bv;
  out[(size_t)(r0 + 2) * D_ + tid] = a2 + bv;
  out[(size_t)(r0 + 3) * D_ + tid] = a3 + bv;
}

extern "C" void kernel_launch(void* const* d_in, const int* in_sizes, int n_in,
                              void* d_out, int out_size, void* d_ws, size_t ws_size,
                              hipStream_t stream) {
  (void)in_sizes; (void)n_in; (void)out_size; (void)ws_size;
  const float* node = (const float*)d_in[0];
  const float* pair = (const float*)d_in[1];
  const int*   mask = (const int*)d_in[2];
  const float* nw   = (const float*)d_in[3];
  const float* nb   = (const float*)d_in[4];
  const float* pw   = (const float*)d_in[5];
  const float* pb   = (const float*)d_in[6];
  const float* Wqkv = (const float*)d_in[7];
  const float* Wb   = (const float*)d_in[8];
  const float* Wout = (const float*)d_in[9];
  const float* bout = (const float*)d_in[10];
  float* out = (float*)d_out;

  char* base = (char*)d_ws;
  size_t off = 0;
  auto take = [&](size_t bytes) -> char* {
    char* p = base + off;
    off = (off + bytes + 255) & ~(size_t)255;
    return p;
  };
  float*    xn   = (float*)take((size_t)N_ * D_ * 4);
  _Float16* qh   = (_Float16*)take((size_t)H_ * N_ * DH_ * 2);
  _Float16* kh   = (_Float16*)take((size_t)H_ * N_ * DH_ * 2);
  _Float16* vT   = (_Float16*)take((size_t)H_ * DH_ * N_ * 2);
  float*    sg   = (float*)take((size_t)H_ * N_ * DH_ * 4);
  _Float16* W2t  = (_Float16*)take((size_t)16 * D_ * 2);
  float*    C0   = (float*)take(256);
  float*    attn = (float*)take((size_t)N_ * INNER_ * 4);
  float*    biasM = (float*)take((size_t)H_ * NN_ * 4);  // total ws ~37.8 MB

  hipLaunchKernelGGL(ln_node_kernel, dim3(N_), dim3(128), 0, stream, node, nw, nb, xn);
  hipLaunchKernelGGL(prep_wbias_kernel, dim3(1), dim3(128), 0, stream, pw, pb, Wb, W2t, C0);
  hipLaunchKernelGGL(qkvg_kernel, dim3(N_ / 4, 4), dim3(256), 0, stream, xn, Wqkv, qh, kh, vT, sg);
  hipLaunchKernelGGL(bias_pair_kernel, dim3(1024), dim3(256), 0, stream, pair, mask, W2t, C0, biasM);
  hipLaunchKernelGGL(attn_kernel, dim3(64), dim3(256), 0, stream, qh, kh, vT, sg, biasM, attn);
  hipLaunchKernelGGL(outproj_kernel, dim3(N_ / 4), dim3(128), 0, stream, attn, Wout, bout, out);
}
